// GNN_Combined_74869869904655
// MI455X (gfx1250) — matrix operand order repaired and checked by reading, not compiled, not run on hardware
//
#include <hip/hip_runtime.h>

#define NS_ 20000
#define ES_ 320000
#define NL_ 50000
#define EL_ 800000
#define IND 256
#define HIDD 128
#define NH 4
#define OUTD 128
#define NCLS 4

typedef __attribute__((ext_vector_type(16))) _Float16 v16h;
typedef __attribute__((ext_vector_type(8)))  float    v8f;

#define CDIV(a, b) (((a) + (b) - 1) / (b))

// ============================================================================
// Fragment packing: convert f32 matrices into WMMA-fragment-ordered f16 so the
// GEMM inner loop is pure coalesced b128 loads + v_wmma.
//
// A fragment (16x32, contiguous k order chosen here, matched by B):
//   lane = m + 16*g ; half t -> k = g*16 + t
// B fragment (32x16):
//   lane = n + 16*g ; half t -> k = g*16 + t
// Packed layout: fragment index ((tile * KB + kb) * 32 + lane), 16 halves each.
// ============================================================================
__global__ void pack_a_f16(const float* __restrict__ A, _Float16* __restrict__ P,
                           int M, int K) {
    int KB = K >> 5;
    int total = (M >> 4) * KB * 32;
    int t = blockIdx.x * blockDim.x + threadIdx.x;
    if (t >= total) return;
    int lane = t & 31;
    int kb = (t >> 5) % KB;
    int tm = (t >> 5) / KB;
    int g = lane >> 4, m = lane & 15;
    const float* ar = A + (size_t)(tm * 16 + m) * K + kb * 32 + g * 16;
    v16h a;
#pragma unroll
    for (int u = 0; u < 8; ++u) {
        a[u]     = (_Float16)ar[u];
        a[u + 8] = (_Float16)ar[u + 8];
    }
    *(v16h*)(P + (size_t)t * 16) = a;
}

__global__ void pack_b_f16(const float* __restrict__ B, _Float16* __restrict__ P,
                           int K, int N) {
    int KB = K >> 5;
    int total = (N >> 4) * KB * 32;
    int t = blockIdx.x * blockDim.x + threadIdx.x;
    if (t >= total) return;
    int lane = t & 31;
    int kb = (t >> 5) % KB;
    int tn = (t >> 5) / KB;
    int g = lane >> 4, n = lane & 15;
    const float* br = B + (size_t)(kb * 32 + g * 16) * N + tn * 16 + n;
    v16h b;
#pragma unroll
    for (int u = 0; u < 16; ++u) b[u] = (_Float16)br[(size_t)u * N];
    *(v16h*)(P + (size_t)t * 16) = b;
}

// ============================================================================
// WMMA GEMM on packed f16 fragments: each wave32 computes a 16x64 strip
// (4 WMMAs per 32-deep K step, A fragment reused 4x). M%16==0, N%64==0, K%32==0.
// ============================================================================
__global__ void gemm_wmma_packed(const _Float16* __restrict__ AP,
                                 const _Float16* __restrict__ BP,
                                 float* __restrict__ C, int M, int K, int N) {
    const int KB = K >> 5;
    const int tilesN4 = N >> 6;                      // groups of 4 n-tiles
    const int wave = blockIdx.x * (blockDim.x >> 5) + (threadIdx.x >> 5);
    const int lane = threadIdx.x & 31;
    const int tiles = (M >> 4) * tilesN4;
    if (wave >= tiles) return;                       // wave-uniform: EXEC all 1s
    const int tm  = wave / tilesN4;
    const int tn4 = wave - tm * tilesN4;

    const v16h* ap  = (const v16h*)AP + (size_t)tm * KB * 32 + lane;
    const v16h* bp0 = (const v16h*)BP + (size_t)(tn4 * 4) * KB * 32 + lane;

    v8f acc0 = {}, acc1 = {}, acc2 = {}, acc3 = {};
    for (int kb = 0; kb < KB; ++kb) {
        v16h a = ap[(size_t)kb * 32];
        if (kb + 1 < KB) __builtin_prefetch(&ap[(size_t)(kb + 1) * 32], 0, 0); // global_prefetch_b8
        v16h b0 = bp0[((size_t)0 * KB + kb) * 32];
        v16h b1 = bp0[((size_t)1 * KB + kb) * 32];
        v16h b2 = bp0[((size_t)2 * KB + kb) * 32];
        v16h b3 = bp0[((size_t)3 * KB + kb) * 32];
        acc0 = __builtin_amdgcn_wmma_f32_16x16x32_f16(false, a, false, b0, (short)0, acc0, false, false);
        acc1 = __builtin_amdgcn_wmma_f32_16x16x32_f16(false, a, false, b1, (short)0, acc1, false, false);
        acc2 = __builtin_amdgcn_wmma_f32_16x16x32_f16(false, a, false, b2, (short)0, acc2, false, false);
        acc3 = __builtin_amdgcn_wmma_f32_16x16x32_f16(false, a, false, b3, (short)0, acc3, false, false);
    }

    const int g = lane >> 4, mn = lane & 15;
    float* crow = C + (size_t)(tm * 16 + g * 8) * N + tn4 * 64 + mn;
#pragma unroll
    for (int r = 0; r < 8; ++r) {
        crow[(size_t)r * N + 0]  = acc0[r];
        crow[(size_t)r * N + 16] = acc1[r];
        crow[(size_t)r * N + 32] = acc2[r];
        crow[(size_t)r * N + 48] = acc3[r];
    }
}

// ---------------- float atomic max via int/uint trick (valid with -inf init)
__device__ inline void atomicMaxFloat(float* addr, float val) {
    if (val >= 0.0f) atomicMax((int*)addr, __float_as_int(val));
    else             atomicMin((unsigned int*)addr, __float_as_uint(val));
}

__device__ inline float lrelu(float x) { return x > 0.0f ? x : 0.2f * x; }

// ---------------- GAT: per (node,head) attention logits; init max=-inf, den=0
__global__ void gat_prep(const float* __restrict__ Hm, const float* __restrict__ al,
                         const float* __restrict__ ar, float* __restrict__ el,
                         float* __restrict__ er, float* __restrict__ mx,
                         float* __restrict__ den, int n_nodes, int heads, int dim) {
    int t = blockIdx.x * blockDim.x + threadIdx.x;
    if (t >= n_nodes * heads) return;
    int h = t % heads;
    const float* hp = Hm + (size_t)t * dim;    // (n*heads+h)*dim
    const float* ap = al + (size_t)h * dim;
    const float* bp = ar + (size_t)h * dim;
    float sl = 0.0f, sr = 0.0f;
    for (int d = 0; d < dim; ++d) { float v = hp[d]; sl += v * ap[d]; sr += v * bp[d]; }
    el[t] = sl; er[t] = sr;
    mx[t] = -INFINITY; den[t] = 0.0f;
}

// ---------------- GAT edge pass 1: segment max of leaky-relu logits over dst
__global__ void gat_edge_max(const int* __restrict__ src, const int* __restrict__ dst,
                             const float* __restrict__ el, const float* __restrict__ er,
                             float* __restrict__ mx, int nE, int heads) {
    int t = blockIdx.x * blockDim.x + threadIdx.x;
    if (t >= nE * heads) return;
    int e = t / heads, h = t - e * heads;
    float v = lrelu(el[src[e] * heads + h] + er[dst[e] * heads + h]);
    atomicMaxFloat(&mx[dst[e] * heads + h], v);
}

// ---------------- GAT edge pass 2: exp weights, denom + weighted feature scatter
// 32 lanes per (edge, head): coalesced atomic sweeps over the feature row.
__global__ void gat_edge_agg(const int* __restrict__ src, const int* __restrict__ dst,
                             const float* __restrict__ el, const float* __restrict__ er,
                             const float* __restrict__ mx, float* __restrict__ den,
                             const float* __restrict__ Hm, float* __restrict__ agg,
                             int nE, int heads, int dim) {
    long long t = (long long)blockIdx.x * blockDim.x + threadIdx.x;
    int lane = (int)(t & 31);
    long long eh = t >> 5;
    if (eh >= (long long)nE * heads) return;
    int e = (int)(eh / heads), h = (int)(eh % heads);
    int si = src[e] * heads + h;
    int di = dst[e] * heads + h;
    float v  = lrelu(el[si] + er[di]);
    float ex = __expf(v - mx[di]);
    if (lane == 0) atomicAdd(&den[di], ex);
    const float* hp = Hm + (size_t)si * dim;
    float* ap = agg + (size_t)di * dim;
    for (int i = lane; i < dim; i += 32) atomicAdd(&ap[i], ex * hp[i]);
}

// ---------------- GAT finalize: agg/den + bias, relu (in place)
__global__ void gat_fin(float* __restrict__ agg, const float* __restrict__ den,
                        const float* __restrict__ b, int n_nodes, int heads, int dim) {
    int t = blockIdx.x * blockDim.x + threadIdx.x;
    int total = n_nodes * heads * dim;
    if (t >= total) return;
    int c  = t % (heads * dim);
    int nh = t / dim;                          // n*heads + h
    float v = agg[t] / fmaxf(den[nh], 1e-9f) + b[c];
    agg[t] = fmaxf(v, 0.0f);
}

// ---------------- GCN degrees
__global__ void deg_acc(const int* __restrict__ src, const int* __restrict__ dst,
                        float* __restrict__ ns, float* __restrict__ nd, int nE) {
    int t = blockIdx.x * blockDim.x + threadIdx.x;
    if (t >= nE) return;
    atomicAdd(&ns[src[t]], 1.0f);
    atomicAdd(&nd[dst[t]], 1.0f);
}
__global__ void deg_norm(float* __restrict__ ns, float* __restrict__ nd, int n) {
    int t = blockIdx.x * blockDim.x + threadIdx.x;
    if (t >= n) return;
    ns[t] = rsqrtf(fmaxf(ns[t], 1.0f));
    nd[t] = rsqrtf(fmaxf(nd[t], 1.0f));
}

// ---------------- GCN scatter: agg[dst] += G[src] * ns[src]; 32 lanes per edge
__global__ void gcn_scatter(const int* __restrict__ src, const int* __restrict__ dst,
                            const float* __restrict__ G, const float* __restrict__ ns,
                            float* __restrict__ agg, int nE, int dim) {
    long long t = (long long)blockIdx.x * blockDim.x + threadIdx.x;
    int lane = (int)(t & 31);
    long long e = t >> 5;
    if (e >= nE) return;
    int s = src[e], d = dst[e];
    float w = ns[s];
    const float* gp = G + (size_t)s * dim;
    float* ap = agg + (size_t)d * dim;
    for (int i = lane; i < dim; i += 32) atomicAdd(&ap[i], w * gp[i]);
}

// ---------------- GCN finalize: agg*nd + bias, optional relu (in place)
__global__ void gcn_fin(float* __restrict__ agg, const float* __restrict__ nd,
                        const float* __restrict__ b, int n, int dim, int do_relu) {
    int t = blockIdx.x * blockDim.x + threadIdx.x;
    if (t >= n * dim) return;
    int nn = t / dim, dd = t - nn * dim;
    float v = agg[t] * nd[nn] + b[dd];
    agg[t] = do_relu ? fmaxf(v, 0.0f) : v;
}

// ---------------- classifier: out[i,c] = [emb_s | emb_l[tok]] . Wcls[:,c] + b[c]
__global__ void classifier(const float* __restrict__ es, const float* __restrict__ elg,
                           const int* __restrict__ tok, const float* __restrict__ W,
                           const float* __restrict__ b, float* __restrict__ out, int n) {
    int t = blockIdx.x * blockDim.x + threadIdx.x;
    if (t >= n * NCLS) return;
    int i = t / NCLS, c = t - i * NCLS;
    const float* a = es  + (size_t)i * OUTD;
    const float* g = elg + (size_t)tok[i] * OUTD;
    float acc = b[c];
    for (int j = 0; j < OUTD; ++j) acc += a[j] * W[j * NCLS + c];
    for (int j = 0; j < OUTD; ++j) acc += g[j] * W[(OUTD + j) * NCLS + c];
    out[t] = acc;
}

// ---------------- helper: pack A+B then run the packed WMMA GEMM
static void run_gemm(const float* A, const float* B, float* C, int M, int K, int N,
                     _Float16* APK, _Float16* BPK, hipStream_t stream) {
    const int Bl = 256;
    int ta = (M >> 4) * (K >> 5) * 32;
    int tb = (N >> 4) * (K >> 5) * 32;
    pack_a_f16<<<CDIV(ta, Bl), Bl, 0, stream>>>(A, APK, M, K);
    pack_b_f16<<<CDIV(tb, Bl), Bl, 0, stream>>>(B, BPK, K, N);
    int waves = (M >> 4) * (N >> 6);
    gemm_wmma_packed<<<CDIV(waves, Bl / 32), Bl, 0, stream>>>(APK, BPK, C, M, K, N);
}

extern "C" void kernel_launch(void* const* d_in, const int* in_sizes, int n_in,
                              void* d_out, int out_size, void* d_ws, size_t ws_size,
                              hipStream_t stream) {
    const float* x_small = (const float*)d_in[0];
    const float* x_large = (const float*)d_in[1];
    const int*   src_s   = (const int*)d_in[2];
    const int*   dst_s   = (const int*)d_in[3];
    const int*   src_l   = (const int*)d_in[4];
    const int*   dst_l   = (const int*)d_in[5];
    const int*   tok     = (const int*)d_in[6];
    const float* Wg1  = (const float*)d_in[7];
    const float* bg1  = (const float*)d_in[8];
    const float* al1  = (const float*)d_in[9];
    const float* ar1  = (const float*)d_in[10];
    const float* Wg2  = (const float*)d_in[11];
    const float* bg2  = (const float*)d_in[12];
    const float* al2  = (const float*)d_in[13];
    const float* ar2  = (const float*)d_in[14];
    const float* Wc1  = (const float*)d_in[15];
    const float* bc1  = (const float*)d_in[16];
    const float* Wc2  = (const float*)d_in[17];
    const float* bc2  = (const float*)d_in[18];
    const float* Wcls = (const float*)d_in[19];
    const float* bcls = (const float*)d_in[20];
    float* out = (float*)d_out;

    // ---- workspace bump allocator
    char* ws = (char*)d_ws;
    size_t off = 0;
    auto allocf = [&](size_t nfloats) -> float* {
        float* p = (float*)(ws + off);
        off += (nfloats * sizeof(float) + 255) & ~(size_t)255;
        return p;
    };
    auto alloch = [&](size_t nhalfs) -> _Float16* {
        _Float16* p = (_Float16*)(ws + off);
        off += (nhalfs * sizeof(_Float16) + 255) & ~(size_t)255;
        return p;
    };
    float* H1    = allocf((size_t)NS_ * NH * HIDD);   // 20000 x 512
    float* AGG1  = allocf((size_t)NS_ * NH * HIDD);
    float* EL1   = allocf((size_t)NS_ * NH);
    float* ER1   = allocf((size_t)NS_ * NH);
    float* M1    = allocf((size_t)NS_ * NH);
    float* DEN1  = allocf((size_t)NS_ * NH);
    float* H2    = allocf((size_t)NS_ * OUTD);        // 20000 x 128
    float* AGG2  = allocf((size_t)NS_ * OUTD);        // -> emb_small
    float* EL2   = allocf((size_t)NS_);
    float* ER2   = allocf((size_t)NS_);
    float* M2    = allocf((size_t)NS_);
    float* DEN2  = allocf((size_t)NS_);
    float* NSRM  = allocf((size_t)NL_);
    float* NDRM  = allocf((size_t)NL_);
    float* G1    = allocf((size_t)NL_ * HIDD);        // 50000 x 128
    float* AGGL1 = allocf((size_t)NL_ * HIDD);        // -> g (relu'd)
    float* G2    = allocf((size_t)NL_ * OUTD);
    float* AGGL2 = allocf((size_t)NL_ * OUTD);        // -> emb_large
    // shared packed-f16 staging (GEMMs run sequentially; sized for the largest)
    _Float16* APK = alloch((size_t)NL_ * IND);        // 12.8M halves (covers all A's)
    _Float16* BPK = alloch((size_t)IND * (NH * HIDD));// 131072 halves (covers all B's)
    (void)ws_size; (void)in_sizes; (void)n_in; (void)out_size;

    const int B = 256;

    // ================= GAT layer 1 (small graph, 4 heads x 128) =================
    run_gemm(x_small, Wg1, H1, NS_, IND, NH * HIDD, APK, BPK, stream);
    hipMemsetAsync(AGG1, 0, (size_t)NS_ * NH * HIDD * sizeof(float), stream);
    gat_prep<<<CDIV(NS_ * NH, B), B, 0, stream>>>(H1, al1, ar1, EL1, ER1, M1, DEN1, NS_, NH, HIDD);
    gat_edge_max<<<CDIV(ES_ * NH, B), B, 0, stream>>>(src_s, dst_s, EL1, ER1, M1, ES_, NH);
    {
        long long th = (long long)ES_ * NH * 32;
        gat_edge_agg<<<(int)CDIV(th, B), B, 0, stream>>>(src_s, dst_s, EL1, ER1, M1, DEN1, H1, AGG1, ES_, NH, HIDD);
    }
    gat_fin<<<CDIV(NS_ * NH * HIDD, B), B, 0, stream>>>(AGG1, DEN1, bg1, NS_, NH, HIDD);

    // ================= GAT layer 2 (small graph, 1 head x 128) =================
    run_gemm(AGG1, Wg2, H2, NS_, NH * HIDD, OUTD, APK, BPK, stream);
    hipMemsetAsync(AGG2, 0, (size_t)NS_ * OUTD * sizeof(float), stream);
    gat_prep<<<CDIV(NS_, B), B, 0, stream>>>(H2, al2, ar2, EL2, ER2, M2, DEN2, NS_, 1, OUTD);
    gat_edge_max<<<CDIV(ES_, B), B, 0, stream>>>(src_s, dst_s, EL2, ER2, M2, ES_, 1);
    {
        long long th = (long long)ES_ * 32;
        gat_edge_agg<<<(int)CDIV(th, B), B, 0, stream>>>(src_s, dst_s, EL2, ER2, M2, DEN2, H2, AGG2, ES_, 1, OUTD);
    }
    gat_fin<<<CDIV(NS_ * OUTD, B), B, 0, stream>>>(AGG2, DEN2, bg2, NS_, 1, OUTD);  // AGG2 = emb_small

    // ================= GCN (large graph) =================
    hipMemsetAsync(NSRM, 0, (size_t)NL_ * sizeof(float), stream);
    hipMemsetAsync(NDRM, 0, (size_t)NL_ * sizeof(float), stream);
    deg_acc<<<CDIV(EL_, B), B, 0, stream>>>(src_l, dst_l, NSRM, NDRM, EL_);
    deg_norm<<<CDIV(NL_, B), B, 0, stream>>>(NSRM, NDRM, NL_);

    run_gemm(x_large, Wc1, G1, NL_, IND, HIDD, APK, BPK, stream);
    hipMemsetAsync(AGGL1, 0, (size_t)NL_ * HIDD * sizeof(float), stream);
    {
        long long th = (long long)EL_ * 32;
        gcn_scatter<<<(int)CDIV(th, B), B, 0, stream>>>(src_l, dst_l, G1, NSRM, AGGL1, EL_, HIDD);
    }
    gcn_fin<<<CDIV(NL_ * HIDD, B), B, 0, stream>>>(AGGL1, NDRM, bc1, NL_, HIDD, 1);  // g = relu(...)

    run_gemm(AGGL1, Wc2, G2, NL_, HIDD, OUTD, APK, BPK, stream);
    hipMemsetAsync(AGGL2, 0, (size_t)NL_ * OUTD * sizeof(float), stream);
    {
        long long th = (long long)EL_ * 32;
        gcn_scatter<<<(int)CDIV(th, B), B, 0, stream>>>(src_l, dst_l, G2, NSRM, AGGL2, EL_, OUTD);
    }
    gcn_fin<<<CDIV(NL_ * OUTD, B), B, 0, stream>>>(AGGL2, NDRM, bc2, NL_, OUTD, 0);  // emb_large

    // ================= classifier =================
    classifier<<<CDIV(NS_ * NCLS, B), B, 0, stream>>>(AGG2, AGGL2, tok, Wcls, bcls, out, NS_);
}
